// HierarchicalManifold_65687229825098
// MI455X (gfx1250) — matrix operand order repaired and checked by reading, not compiled, not run on hardware
//
#include <hip/hip_runtime.h>
#include <hip/hip_bf16.h>
#include <math.h>

typedef __attribute__((ext_vector_type(16))) __bf16 v16bf;
typedef __attribute__((ext_vector_type(8)))  __bf16 v8bf;
typedef __attribute__((ext_vector_type(8)))  float  v8f;

#define NT     256
#define NWAVES 8

__device__ __forceinline__ v8f wmma_bf16(v16bf a, v16bf b, v8f c) {
  return __builtin_amdgcn_wmma_f32_16x16x32_bf16(false, a, false, b, (short)0, c, false, false);
}

// A fragment from row-major (M x K) bf16, ldk elements per row.
// CDNA5 A layout: lane m=lane&15, h=lane>>4; elems 0..7 -> k=8h+e, elems 8..15 -> k=16+8h+e.
__device__ __forceinline__ v16bf frag_a(const __bf16* p, int mBase, int ldk, int lane) {
  const __bf16* q = p + (mBase + (lane & 15)) * ldk + 8 * ((lane >> 4) & 1);
  v8bf lo = *(const v8bf*)(q);
  v8bf hi = *(const v8bf*)(q + 16);
  v16bf r;
#pragma unroll
  for (int i = 0; i < 8; ++i) { r[i] = lo[i]; r[8 + i] = hi[i]; }
  return r;
}

// B fragment from TRANSPOSED storage (N x K row-major) bf16.
// CDNA5 B layout: lane n=lane&15, h=lane>>4; elems 0..7 -> k=8h+e, elems 8..15 -> k=16+8h+e.
__device__ __forceinline__ v16bf frag_b(const __bf16* p, int nBase, int ldk, int lane) {
  const __bf16* q = p + (nBase + (lane & 15)) * ldk + 8 * ((lane >> 4) & 1);
  v8bf lo = *(const v8bf*)(q);
  v8bf hi = *(const v8bf*)(q + 16);
  v16bf r;
#pragma unroll
  for (int i = 0; i < 8; ++i) { r[i] = lo[i]; r[8 + i] = hi[i]; }
  return r;
}

// One automata step. S: N x 32 f32. bA: row-major bf16 mirror. bT: transposed
// (32 x TK) bf16 mirror (zero-padded past k=N when TK>N). Tr: 32x32 f32.
template <int N>
__device__ __forceinline__ void automata(
    float* S, __bf16* bA, __bf16* bT, float* Tr, __bf16* bTrT,
    float* sNewF, __bf16* bNewA, __bf16* bNewT, __bf16* bAgg,
    const __bf16* wA, const __bf16* wWsT, const __bf16* wWaT,
    const float* __restrict__ bias, float* sVec, int wave, int lane, int tid) {
  constexpr int TK = (N == 64) ? 64 : 32;
  constexpr int MT = N / 16;
  constexpr int KCH = TK / 32;
  // agg = A(NxN) @ S(Nx32) -> bAgg (bf16, row-major)
  for (int t = wave; t < MT * 2; t += NWAVES) {
    int mB = (t >> 1) * 16, nB = (t & 1) * 16;
    v8f acc = {};
#pragma unroll
    for (int c = 0; c < KCH; ++c)
      acc = wmma_bf16(frag_a(wA + c * 32, mB, TK, lane),
                      frag_b(bT + c * 32, nB, TK, lane), acc);
    int col = nB + (lane & 15), hh = (lane >> 4) & 1;
#pragma unroll
    for (int r = 0; r < 8; ++r)
      bAgg[(mB + r + 8 * hh) * 32 + col] = (__bf16)acc[r];
  }
  __syncthreads();
  // upd = tanh(S@Tr + S@Ws + agg@Wa + b); sNewF = S + 0.1*upd
  for (int t = wave; t < MT * 2; t += NWAVES) {
    int mB = (t >> 1) * 16, nB = (t & 1) * 16;
    v8f acc = {};
    v16bf aS = frag_a(bA, mB, 32, lane);
    acc = wmma_bf16(aS, frag_b(bTrT, nB, 32, lane), acc);
    acc = wmma_bf16(aS, frag_b(wWsT, nB, 32, lane), acc);
    acc = wmma_bf16(frag_a(bAgg, mB, 32, lane), frag_b(wWaT, nB, 32, lane), acc);
    int col = nB + (lane & 15), hh = (lane >> 4) & 1;
    float bcol = bias[col];
#pragma unroll
    for (int r = 0; r < 8; ++r) {
      int m = mB + r + 8 * hh;
      sNewF[m * 32 + col] = S[m * 32 + col] + 0.1f * tanhf(acc[r] + bcol);
    }
  }
  __syncthreads();
  // l2 row norms
  if (tid < N) {
    float ss = 0.f;
#pragma unroll
    for (int c = 0; c < 32; ++c) { float x = sNewF[tid * 32 + c]; ss += x * x; }
    sVec[tid] = 1.0f / (sqrtf(ss) + 1e-6f);
  }
  __syncthreads();
  for (int i = tid; i < N * 32; i += NT) {
    float v = sNewF[i] * sVec[i >> 5];
    sNewF[i] = v;
    bNewA[i] = (__bf16)v;
    bNewT[(i & 31) * TK + (i >> 5)] = (__bf16)v;
  }
  if (TK > N) {  // zero-pad k in [N, TK) of transposed new-state
    for (int i = tid; i < 32 * (TK - N); i += NT)
      bNewT[(i >> 4) * TK + N + (i & 15)] = (__bf16)0.0f;
  }
  __syncthreads();
  // Tr = 0.95*Tr + (0.05/N) * S^T @ Snew  (A = OLD bT, B = bNewT)
  const float tsc = 0.05f / (float)N;
  for (int t = wave; t < 4; t += NWAVES) {
    int mB = (t >> 1) * 16, nB = (t & 1) * 16;
    v8f acc = {};
#pragma unroll
    for (int c = 0; c < KCH; ++c)
      acc = wmma_bf16(frag_a(bT + c * 32, mB, TK, lane),
                      frag_b(bNewT + c * 32, nB, TK, lane), acc);
    int col = nB + (lane & 15), hh = (lane >> 4) & 1;
#pragma unroll
    for (int r = 0; r < 8; ++r) {
      int m = mB + r + 8 * hh;
      Tr[m * 32 + col] = 0.95f * Tr[m * 32 + col] + tsc * acc[r];
    }
  }
  __syncthreads();
  // commit: S = Snew (f32 + bf16 mirrors), refresh bf16 transposed trace
  for (int i = tid; i < N * 32; i += NT) { S[i] = sNewF[i]; bA[i] = bNewA[i]; }
  for (int i = tid; i < 32 * TK; i += NT) bT[i] = bNewT[i];
  for (int i = tid; i < 1024; i += NT)
    bTrT[i] = (__bf16)Tr[(i & 31) * 32 + (i >> 5)];
  __syncthreads();
}

__global__ void __launch_bounds__(NT)
hm_kernel(const float* __restrict__ micro_in, const float* __restrict__ macro_in,
          const float* __restrict__ trmi_in, const float* __restrict__ trma_in,
          const float* __restrict__ W_td, const float* __restrict__ b_td,
          const float* __restrict__ ln_g, const float* __restrict__ ln_b,
          const float* __restrict__ Wqkv, const float* __restrict__ bqkv,
          const float* __restrict__ Wo, const float* __restrict__ bo,
          const float* __restrict__ A_mi, const float* __restrict__ Ws_mi,
          const float* __restrict__ Wa_mi, const float* __restrict__ b_mi,
          const float* __restrict__ A_ma, const float* __restrict__ Ws_ma,
          const float* __restrict__ Wa_ma, const float* __restrict__ b_ma,
          const float* __restrict__ We1, const float* __restrict__ be1,
          const float* __restrict__ We2, const float* __restrict__ be2,
          const int* __restrict__ steps_p,
          float* __restrict__ out_micro, float* __restrict__ out_macro,
          float* __restrict__ out_trmi, float* __restrict__ out_trma,
          float* __restrict__ out_energy) {
  extern __shared__ char smem[];
  float* sMicro = (float*)smem;          // 2048
  float* sMacro = sMicro + 2048;         // 512
  float* sTrMi  = sMacro + 512;          // 1024
  float* sTrMa  = sTrMi + 1024;          // 1024
  float* fScr   = sTrMa + 1024;          // 4096 (sNew | SC+RES | energy-h)
  float* sVec   = fScr + 4096;           // 128
  __bf16* bb = (__bf16*)(sVec + 128);
  __bf16* bSA    = bb;             // 2048  micro row-major
  __bf16* bST    = bb + 2048;      // 2048  micro^T (32x64)
  __bf16* bMacA  = bb + 4096;      // 512   macro row-major
  __bf16* bMacT  = bb + 4608;      // 1024  macro^T padded (32x32, k>=16 zero)
  __bf16* bTrMiT = bb + 5632;      // 1024
  __bf16* bTrMaT = bb + 6656;      // 1024
  __bf16* bNewA  = bb + 7680;      // 2048
  __bf16* bNewT  = bb + 9728;      // 2048 (32x64 / 32x32 prefix)
  __bf16* bAgg   = bb + 11776;     // 2048
  __bf16* bQ     = bb + 13824;     // 1024  2 x (16x32), other head zeroed
  __bf16* bKA    = bb + 14848;     // 2048  kp row-major (64x32)
  __bf16* bVT    = bb + 16896;     // 2048  vp^T (32x64)
  __bf16* bAttn  = bb + 18944;     // 2048  2 x (16x64) row-major
  __bf16* bOut   = bb + 20992;     // 512   16x32 row-major
  __bf16* wAmi   = bb + 21504;     // 4096  A_mi row-major 64x64
  __bf16* wWsMiT = bb + 25600;     // 1024  Ws_mi^T
  __bf16* wWaMiT = bb + 26624;     // 1024  Wa_mi^T
  __bf16* wAma   = bb + 27648;     // 512   A_ma padded 16x32
  __bf16* wWsMaT = bb + 28160;     // 1024
  __bf16* wWaMaT = bb + 29184;     // 1024
  __bf16* wQkvT  = bb + 30208;     // 3072  Wqkv native (out,in) == B^T layout
  __bf16* wWoN   = bb + 33280;     // 1024  Wo native
  __bf16* wWe1T  = bb + 34304;     // 2048  We1^T (64x32)

  const int tid = threadIdx.x;
  const int lane = tid & 31;
  const int wave = tid >> 5;
  const int b = blockIdx.x;

  // ---- one-time per-block init: states + bf16 weight layouts ----
  for (int i = tid; i < 2048; i += NT) sMicro[i] = micro_in[b * 2048 + i];
  for (int i = tid; i < 512; i += NT) {
    float v = macro_in[b * 512 + i];
    sMacro[i] = v; bMacA[i] = (__bf16)v;
  }
  for (int i = tid; i < 1024; i += NT) {  // bMacT[n*32+k] = k<16 ? macro[k][n] : 0
    int n = i >> 5, k = i & 31;
    bMacT[i] = (k < 16) ? (__bf16)macro_in[b * 512 + k * 32 + n] : (__bf16)0.0f;
  }
  for (int i = tid; i < 1024; i += NT) {
    float v = trmi_in[b * 1024 + i];
    sTrMi[i] = v;
    bTrMiT[(i & 31) * 32 + (i >> 5)] = (__bf16)v;
  }
  for (int i = tid; i < 1024; i += NT) {
    float v = trma_in[b * 1024 + i];
    sTrMa[i] = v;
    bTrMaT[(i & 31) * 32 + (i >> 5)] = (__bf16)v;
  }
  for (int i = tid; i < 4096; i += NT) wAmi[i] = (__bf16)A_mi[i];
  for (int i = tid; i < 1024; i += NT) {
    int j = i >> 5, k = i & 31;
    wWsMiT[i] = (__bf16)Ws_mi[k * 32 + j];
    wWaMiT[i] = (__bf16)Wa_mi[k * 32 + j];
    wWsMaT[i] = (__bf16)Ws_ma[k * 32 + j];
    wWaMaT[i] = (__bf16)Wa_ma[k * 32 + j];
    wWoN[i] = (__bf16)Wo[i];
  }
  for (int i = tid; i < 512; i += NT) {
    int m = i >> 5, k = i & 31;
    wAma[i] = (k < 16) ? (__bf16)A_ma[m * 16 + k] : (__bf16)0.0f;
  }
  for (int i = tid; i < 3072; i += NT) wQkvT[i] = (__bf16)Wqkv[i];
  for (int i = tid; i < 2048; i += NT) {
    int j = i >> 5, k = i & 31;
    wWe1T[i] = (__bf16)We1[k * 64 + j];
  }
  __syncthreads();

  const int S = steps_p[0];
  for (int s = 0; s < S; ++s) {
    // ---- top-down bias: micro += 0.1*(mean(macro) @ W_td^T + b_td) ----
    if (tid < 32) {
      float m = 0.f;
#pragma unroll
      for (int r = 0; r < 16; ++r) m += sMacro[r * 32 + tid];
      sVec[tid] = m * (1.0f / 16.0f);
    }
    __syncthreads();
    if (tid < 32) {
      float acc = b_td[tid];
#pragma unroll
      for (int k = 0; k < 32; ++k) acc += sVec[k] * W_td[tid * 32 + k];
      sVec[64 + tid] = 0.1f * acc;
    }
    __syncthreads();
    for (int i = tid; i < 2048; i += NT) {
      float v = sMicro[i] + sVec[64 + (i & 31)];
      sMicro[i] = v;
      bSA[i] = (__bf16)v;
      bST[(i & 31) * 64 + (i >> 5)] = (__bf16)v;
    }
    __syncthreads();

    // ---- micro automata (N=64) ----
    automata<64>(sMicro, bSA, bST, sTrMi, bTrMiT, fScr, bNewA, bNewT, bAgg,
                 wAmi, wWsMiT, wWaMiT, b_mi, sVec, wave, lane, tid);

    // ---- MHA ----
    float* SC = fScr;          // 2 x 16 x 64 f32 scores
    float* RES = fScr + 2560;  // 16 x 32 f32 residual
    // qp = macro @ Wq^T + bq  -> bQ (per-head, other head zeroed)
    for (int t = wave; t < 2; t += NWAVES) {
      v8f acc = wmma_bf16(frag_a(bMacA, 0, 32, lane),
                          frag_b(wQkvT, t * 16, 32, lane), (v8f){});
      int col = t * 16 + (lane & 15), hh = (lane >> 4) & 1;
      int hd = col >> 4;
      float bq = bqkv[col];
#pragma unroll
      for (int r = 0; r < 8; ++r) {
        int m = r + 8 * hh;
        bQ[hd * 512 + m * 32 + col] = (__bf16)(acc[r] + bq);
        bQ[(1 - hd) * 512 + m * 32 + col] = (__bf16)0.0f;
      }
    }
    // kp, vp from micro -> bKA (row-major), bVT (transposed)
    for (int t = wave; t < 8; t += NWAVES) {
      int mB = (t >> 1) * 16, nB = (t & 1) * 16;
      v16bf aM = frag_a(bSA, mB, 32, lane);
      v8f ak = wmma_bf16(aM, frag_b(wQkvT + 1024, nB, 32, lane), (v8f){});
      v8f av = wmma_bf16(aM, frag_b(wQkvT + 2048, nB, 32, lane), (v8f){});
      int col = nB + (lane & 15), hh = (lane >> 4) & 1;
      float bk = bqkv[32 + col], bv = bqkv[64 + col];
#pragma unroll
      for (int r = 0; r < 8; ++r) {
        int m = mB + r + 8 * hh;
        bKA[m * 32 + col] = (__bf16)(ak[r] + bk);
        bVT[col * 64 + m] = (__bf16)(av[r] + bv);
      }
    }
    __syncthreads();
    // scores_h = (q_h_padded @ kp^T) * 0.25  (zeros in A isolate the head)
    for (int t = wave; t < 8; t += NWAVES) {
      int h = t >> 2, nt = t & 3;
      v8f acc = wmma_bf16(frag_a(bQ + h * 512, 0, 32, lane),
                          frag_b(bKA, nt * 16, 32, lane), (v8f){});
      int col = nt * 16 + (lane & 15), hh = (lane >> 4) & 1;
#pragma unroll
      for (int r = 0; r < 8; ++r)
        SC[h * 1024 + (r + 8 * hh) * 64 + col] = acc[r] * 0.25f;
    }
    __syncthreads();
    // softmax rows -> bAttn (bf16)
    if (tid < 32) {
      float* row = SC + (tid >> 4) * 1024 + (tid & 15) * 64;
      __bf16* arow = bAttn + (tid >> 4) * 1024 + (tid & 15) * 64;
      float mx = row[0];
#pragma unroll
      for (int k = 1; k < 64; ++k) mx = fmaxf(mx, row[k]);
      float sum = 0.f;
      float e[64];
#pragma unroll
      for (int k = 0; k < 64; ++k) { e[k] = expf(row[k] - mx); sum += e[k]; }
      float inv = 1.0f / sum;
#pragma unroll
      for (int k = 0; k < 64; ++k) arow[k] = (__bf16)(e[k] * inv);
    }
    __syncthreads();
    // out_h = attn_h @ v_h -> bOut
    for (int t = wave; t < 2; t += NWAVES) {
      v8f acc = {};
#pragma unroll
      for (int k0 = 0; k0 < 64; k0 += 32)
        acc = wmma_bf16(frag_a(bAttn + t * 1024 + k0, 0, 64, lane),
                        frag_b(bVT + t * 16 * 64 + k0, 0, 64, lane), acc);
      int col = t * 16 + (lane & 15), hh = (lane >> 4) & 1;
#pragma unroll
      for (int r = 0; r < 8; ++r)
        bOut[(r + 8 * hh) * 32 + col] = (__bf16)acc[r];
    }
    __syncthreads();
    // integrated = out @ Wo^T + bo ; RES = macro + integrated
    for (int t = wave; t < 2; t += NWAVES) {
      v8f acc = wmma_bf16(frag_a(bOut, 0, 32, lane),
                          frag_b(wWoN, t * 16, 32, lane), (v8f){});
      int col = t * 16 + (lane & 15), hh = (lane >> 4) & 1;
      float bc = bo[col];
#pragma unroll
      for (int r = 0; r < 8; ++r) {
        int m = r + 8 * hh;
        RES[m * 32 + col] = sMacro[m * 32 + col] + acc[r] + bc;
      }
    }
    __syncthreads();
    // layernorm -> sMacro (+ bf16 mirrors)
    if (tid < 16) {
      float mu = 0.f;
#pragma unroll
      for (int c = 0; c < 32; ++c) mu += RES[tid * 32 + c];
      mu *= (1.0f / 32.0f);
      float var = 0.f;
#pragma unroll
      for (int c = 0; c < 32; ++c) { float d = RES[tid * 32 + c] - mu; var += d * d; }
      sVec[tid] = mu;
      sVec[32 + tid] = 1.0f / sqrtf(var * (1.0f / 32.0f) + 1e-5f);
    }
    __syncthreads();
    for (int i = tid; i < 512; i += NT) {
      int r = i >> 5, c = i & 31;
      float v = (RES[i] - sVec[r]) * sVec[32 + r] * ln_g[c] + ln_b[c];
      sMacro[i] = v;
      bMacA[i] = (__bf16)v;
      bMacT[c * 32 + r] = (__bf16)v;  // k=r<16; pad k>=16 stays zero from init
    }
    __syncthreads();

    // ---- macro automata (N=16) ----
    automata<16>(sMacro, bMacA, bMacT, sTrMa, bTrMaT, fScr, bNewA, bNewT, bAgg,
                 wAma, wWsMaT, wWaMaT, b_ma, sVec, wave, lane, tid);

    // ---- energy (last step only) ----
    if (s == S - 1) {
      for (int t = wave; t < 16; t += NWAVES) {
        int mB = (t >> 2) * 16, nB = (t & 3) * 16;
        v8f acc = wmma_bf16(frag_a(bSA, mB, 32, lane),
                            frag_b(wWe1T, nB, 32, lane), (v8f){});
        int col = nB + (lane & 15), hh = (lane >> 4) & 1;
        float bc = be1[col];
#pragma unroll
        for (int r = 0; r < 8; ++r)
          fScr[(mB + r + 8 * hh) * 64 + col] = tanhf(acc[r] + bc);
      }
      __syncthreads();
      if (tid < 64) {
        float d = be2[0];
#pragma unroll
        for (int k = 0; k < 64; ++k) d += fScr[tid * 64 + k] * We2[k];
        sVec[tid] = d;
      }
      __syncthreads();
      if (tid == 0) {
        float e = 0.f;
#pragma unroll
        for (int k = 0; k < 64; ++k) e += sVec[k];
        out_energy[b] = e * (1.0f / 64.0f);
      }
      __syncthreads();
    }
  }

  // ---- write back ----
  for (int i = tid; i < 2048; i += NT) out_micro[b * 2048 + i] = sMicro[i];
  for (int i = tid; i < 512;  i += NT) out_macro[b * 512 + i]  = sMacro[i];
  for (int i = tid; i < 1024; i += NT) out_trmi[b * 1024 + i]  = sTrMi[i];
  for (int i = tid; i < 1024; i += NT) out_trma[b * 1024 + i]  = sTrMa[i];
}

extern "C" void kernel_launch(void* const* d_in, const int* in_sizes, int n_in,
                              void* d_out, int out_size, void* d_ws, size_t ws_size,
                              hipStream_t stream) {
  const float* micro_in = (const float*)d_in[0];
  const float* macro_in = (const float*)d_in[1];
  const float* trmi_in  = (const float*)d_in[2];
  const float* trma_in  = (const float*)d_in[3];
  const float* W_td = (const float*)d_in[4];
  const float* b_td = (const float*)d_in[5];
  const float* ln_g = (const float*)d_in[6];
  const float* ln_b = (const float*)d_in[7];
  const float* Wqkv = (const float*)d_in[8];
  const float* bqkv = (const float*)d_in[9];
  const float* Wo   = (const float*)d_in[10];
  const float* bo   = (const float*)d_in[11];
  const float* A_mi = (const float*)d_in[12];
  const float* Ws_mi = (const float*)d_in[13];
  const float* Wa_mi = (const float*)d_in[14];
  const float* b_mi  = (const float*)d_in[15];
  const float* A_ma  = (const float*)d_in[16];
  const float* Ws_ma = (const float*)d_in[17];
  const float* Wa_ma = (const float*)d_in[18];
  const float* b_ma  = (const float*)d_in[19];
  const float* We1 = (const float*)d_in[20];
  const float* be1 = (const float*)d_in[21];
  const float* We2 = (const float*)d_in[22];
  const float* be2 = (const float*)d_in[23];
  const int* steps = (const int*)d_in[24];

  const int Bn = in_sizes[0] / (64 * 32);
  float* out = (float*)d_out;
  float* out_micro = out;
  float* out_macro = out_micro + (size_t)Bn * 2048;
  float* out_trmi  = out_macro + (size_t)Bn * 512;
  float* out_trma  = out_trmi + (size_t)Bn * 1024;
  float* out_energy = out_trma + (size_t)Bn * 1024;

  // f32: (2048+512+1024+1024+4096+128)*4 = 35328 B ; bf16: 36352*2 = 72704 B
  const size_t shmem = 35328 + 72704;  // 108032 B (< 320 KB/WG on CDNA5)

  hm_kernel<<<Bn, NT, shmem, stream>>>(
      micro_in, macro_in, trmi_in, trma_in, W_td, b_td, ln_g, ln_b, Wqkv, bqkv,
      Wo, bo, A_mi, Ws_mi, Wa_mi, b_mi, A_ma, Ws_ma, Wa_ma, b_ma, We1, be1,
      We2, be2, steps, out_micro, out_macro, out_trmi, out_trma, out_energy);
}
